// CCA_SSG_41824391528816
// MI455X (gfx1250) — matrix operand order repaired and checked by reading, not compiled, not run on hardware
//
#include <hip/hip_runtime.h>
#include <math.h>

typedef __bf16 bf16;
typedef __attribute__((ext_vector_type(16))) __bf16 v16bf;
typedef __attribute__((ext_vector_type(8)))  float  v8f;
typedef __attribute__((ext_vector_type(4)))  float  f4;

#define G_ 64
#define N_ 400
#define E_ 12800
#define F_ 400
#define C_ 128
#define H_ 512

#define BM 64
#define BN 128
#define BK 32
#define APITCH 40   // bf16 elems; 80B row pitch: 16B aligned, conflict-free

// ---------- helpers ----------
__device__ __forceinline__ unsigned f2ord(float f) {
  unsigned u = __float_as_uint(f);
  return (u & 0x80000000u) ? ~u : (u | 0x80000000u);
}
__device__ __forceinline__ float ord2f(unsigned u) {
  return (u & 0x80000000u) ? __uint_as_float(u & 0x7fffffffu) : __uint_as_float(~u);
}

// ---------- WMMA bf16 GEMM with async global->LDS double buffering ----------
// Out[z] = act(A[z][M,K] @ B[K,N] + bias). Block: 256 thr = 8 waves, tile 64x128.
// Each wave: 32x32 (2x2 wmma). fp32 tiles stream in via GLOBAL_LOAD_ASYNC_TO_LDS,
// a cooperative pass converts to bf16 (B transposed) for 16B-aligned ds fragment loads.
__global__ __launch_bounds__(256) void k_gemm_bf16(
    const float* __restrict__ A, const float* __restrict__ B,
    const float* __restrict__ bias, float* __restrict__ Out,
    int M, int K, int N, long long strideA, long long strideO, int relu) {
  __shared__ float Af32[2][BM][BK];       // 16 KB (double-buffered async dest)
  __shared__ float Bf32[2][BK][BN];       // 32 KB
  __shared__ bf16  Abf[BM][APITCH];       // 5 KB  ([m][k])
  __shared__ bf16  Btr[BN][APITCH];       // 10 KB ([n][k], transposed)

  const int t     = threadIdx.x;
  const int m0blk = blockIdx.x * BM;
  const int n0blk = blockIdx.y * BN;
  const float* Ab = A + (long long)blockIdx.z * strideA;
  float*       Ob = Out + (long long)blockIdx.z * strideO;

  // ---- async stage of one k-block (fp32, raw copy; OOB chunks zero-filled) ----
  auto issueA = [&](int buf, int k0) {
#pragma unroll
    for (int i = 0; i < 2; ++i) {
      const int q   = t + i * 256;        // 0..511 float4 chunks (64x32 fp32)
      const int row = q >> 3;
      const int fc  = (q & 7) * 4;
      float* dst = &Af32[buf][row][fc];
      const int gm = m0blk + row, gk = k0 + fc;
      if (gm < M && gk + 3 < K) {
        const float* src = Ab + (long long)gm * K + gk;
        asm volatile("global_load_async_to_lds_b128 %0, %1, off"
                     :: "v"((unsigned)(uintptr_t)dst),
                        "v"((unsigned long long)(uintptr_t)src)
                     : "memory");
      } else {
        *(f4*)dst = (f4){0.f, 0.f, 0.f, 0.f};
      }
    }
  };
  auto issueB = [&](int buf, int k0) {
#pragma unroll
    for (int i = 0; i < 4; ++i) {
      const int q  = t + i * 256;         // 0..1023 float4 chunks (32x128 fp32)
      const int kr = q >> 5;
      const int fc = (q & 31) * 4;
      float* dst = &Bf32[buf][kr][fc];
      const int gk = k0 + kr;
      if (gk < K) {
        const float* src = B + (long long)gk * N + n0blk + fc;
        asm volatile("global_load_async_to_lds_b128 %0, %1, off"
                     :: "v"((unsigned)(uintptr_t)dst),
                        "v"((unsigned long long)(uintptr_t)src)
                     : "memory");
      } else {
        *(f4*)dst = (f4){0.f, 0.f, 0.f, 0.f};
      }
    }
  };
  // ---- cooperative fp32 -> bf16 conversion (B transposed to [n][k]) ----
  auto convert = [&](int buf) {
    {
      const int row = t >> 2, c = (t & 3) * 8;
      f4 x0 = *(f4*)&Af32[buf][row][c];
      f4 x1 = *(f4*)&Af32[buf][row][c + 4];
      bf16* d = &Abf[row][c];
#pragma unroll
      for (int j = 0; j < 4; ++j) d[j] = (bf16)x0[j];
#pragma unroll
      for (int j = 0; j < 4; ++j) d[4 + j] = (bf16)x1[j];
    }
    {
      const int n = t & 127, kh = (t >> 7) * 16;
      bf16* d = &Btr[n][kh];
#pragma unroll
      for (int j = 0; j < 16; ++j) d[j] = (bf16)Bf32[buf][kh + j][n];
    }
  };

  const int w = t >> 5, lid = t & 31;
  const int half = lid >> 4, l16 = lid & 15;
  const int wm = (w & 1) * 32, wn = (w >> 1) * 32;

  v8f acc[2][2] = {};
  int buf = 0;
  issueA(0, 0);
  issueB(0, 0);
  for (int k0 = 0; k0 < K; k0 += BK) {
    asm volatile("s_wait_asynccnt 0" ::: "memory");
    __syncthreads();                       // tile[buf] resident & visible
    if (k0 + BK < K) { issueA(buf ^ 1, k0 + BK); issueB(buf ^ 1, k0 + BK); }
    convert(buf);
    __syncthreads();                       // bf16 panels ready

    v16bf af[2], bfr[2];
#pragma unroll
    for (int mi = 0; mi < 2; ++mi) {
      const bf16* ap = &Abf[wm + mi * 16 + l16][0];
#pragma unroll
      for (int j = 0; j < 8; ++j) af[mi][j] = ap[half * 16 + j];
#pragma unroll
      for (int j = 0; j < 8; ++j) af[mi][8 + j] = ap[half * 16 + 8 + j];
    }
#pragma unroll
    for (int ni = 0; ni < 2; ++ni) {
      const bf16* bp = &Btr[wn + ni * 16 + l16][half * 16];
#pragma unroll
      for (int j = 0; j < 16; ++j) bfr[ni][j] = bp[j];
    }
    acc[0][0] = __builtin_amdgcn_wmma_f32_16x16x32_bf16(false, af[0], false, bfr[0], (short)0, acc[0][0], false, false);
    acc[0][1] = __builtin_amdgcn_wmma_f32_16x16x32_bf16(false, af[0], false, bfr[1], (short)0, acc[0][1], false, false);
    acc[1][0] = __builtin_amdgcn_wmma_f32_16x16x32_bf16(false, af[1], false, bfr[0], (short)0, acc[1][0], false, false);
    acc[1][1] = __builtin_amdgcn_wmma_f32_16x16x32_bf16(false, af[1], false, bfr[1], (short)0, acc[1][1], false, false);
    __syncthreads();                       // fragments consumed; panels reusable
    buf ^= 1;
  }

  // ---- epilogue: element r -> row (half*8 + r), col l16 of each 16x16 tile ----
#pragma unroll
  for (int mi = 0; mi < 2; ++mi) {
#pragma unroll
    for (int ni = 0; ni < 2; ++ni) {
      const int n = n0blk + wn + ni * 16 + l16;
      const float bv = bias ? bias[n] : 0.f;
#pragma unroll
      for (int r = 0; r < 8; ++r) {
        const int m = m0blk + wm + mi * 16 + half * 8 + r;
        if (m < M) {
          float v = acc[mi][ni][r] + bv;
          if (relu) v = fmaxf(v, 0.f);
          Ob[(long long)m * N + n] = v;
        }
      }
    }
  }
}

// ---------- tiny dot: wedot = dot(We[1xC], a_edge[C]) ----------
__global__ void k_wedot(const float* __restrict__ We, const float* __restrict__ ae,
                        float* __restrict__ out, int idx) {
  __shared__ float s[C_];
  int c = threadIdx.x;
  s[c] = We[c] * ae[c];
  __syncthreads();
  for (int off = 64; off > 0; off >>= 1) {
    if (c < off) s[c] += s[c + off];
    __syncthreads();
  }
  if (c == 0) out[idx] = s[0];
}

// ---------- per-node prep: alpha dots, zero accumulators ----------
__global__ void k_node_prep(const float* __restrict__ XP,
                            const float* __restrict__ a_src, const float* __restrict__ a_dst,
                            float* __restrict__ aS, float* __restrict__ aD,
                            float* __restrict__ cnt, float* __restrict__ easum,
                            unsigned* __restrict__ amax, float* __restrict__ den,
                            float* __restrict__ Hb) {
  const int n = blockIdx.x, z = blockIdx.y, c = threadIdx.x;
  const long long ni = (long long)z * N_ + n;
  const float x = XP[ni * C_ + c];
  __shared__ float s1[C_], s2[C_];
  s1[c] = x * a_src[c];
  s2[c] = x * a_dst[c];
  __syncthreads();
  for (int off = 64; off > 0; off >>= 1) {
    if (c < off) { s1[c] += s1[c + off]; s2[c] += s2[c + off]; }
    __syncthreads();
  }
  if (c == 0) {
    aS[ni] = s1[0]; aD[ni] = s2[0];
    cnt[ni] = 0.f; easum[ni] = 0.f; den[ni] = 0.f;
    amax[ni] = 0x007FFFFFu;            // f2ord(-inf)
  }
  Hb[ni * C_ + c] = 0.f;
}

// ---------- edge stats for self-loop fill_value='mean' ----------
__global__ void k_edge_stats(const int* __restrict__ EI, const float* __restrict__ EA,
                             float* __restrict__ cnt, float* __restrict__ easum, int g0) {
  const int e = blockIdx.x * blockDim.x + threadIdx.x;
  const int z = blockIdx.y;
  if (e >= E_) return;
  const int g = g0 + z;
  const int d = EI[(long long)g * 2 * E_ + E_ + e];
  atomicAdd(&cnt[(long long)z * N_ + d], 1.f);
  atomicAdd(&easum[(long long)z * N_ + d], EA[(long long)g * E_ + e]);
}

// ---------- attention logits + segment max ----------
__global__ void k_edge_logit(const int* __restrict__ EI, const float* __restrict__ EA,
                             const float* __restrict__ aS, const float* __restrict__ aD,
                             const float* __restrict__ cnt, const float* __restrict__ easum,
                             const float* __restrict__ wedot, int wi,
                             float* __restrict__ Abuf, unsigned* __restrict__ amax, int g0) {
  const int i = blockIdx.x * blockDim.x + threadIdx.x;
  const int z = blockIdx.y;
  if (i >= E_ + N_) return;
  const int g = g0 + z;
  const long long nb = (long long)z * N_;
  int s, d; float eav;
  if (i < E_) {
    s = EI[(long long)g * 2 * E_ + i];
    d = EI[(long long)g * 2 * E_ + E_ + i];
    eav = EA[(long long)g * E_ + i];
  } else {
    s = d = i - E_;
    eav = easum[nb + d] / fmaxf(cnt[nb + d], 1.f);
  }
  float a = aS[nb + s] + aD[nb + d] + eav * wedot[wi];
  a = (a > 0.f) ? a : 0.2f * a;                      // leaky_relu 0.2
  Abuf[(long long)z * (E_ + N_) + i] = a;
  atomicMax(&amax[nb + d], f2ord(a));
}

// ---------- exp + segment sum ----------
__global__ void k_edge_expsum(const int* __restrict__ EI, float* __restrict__ Abuf,
                              const unsigned* __restrict__ amax, float* __restrict__ den,
                              int g0) {
  const int i = blockIdx.x * blockDim.x + threadIdx.x;
  const int z = blockIdx.y;
  if (i >= E_ + N_) return;
  const int g = g0 + z;
  const int d = (i < E_) ? EI[(long long)g * 2 * E_ + E_ + i] : (i - E_);
  const long long nb = (long long)z * N_;
  const long long zb = (long long)z * (E_ + N_);
  float ex = __expf(Abuf[zb + i] - ord2f(amax[nb + d]));
  Abuf[zb + i] = ex;
  atomicAdd(&den[nb + d], ex);
}

// ---------- weighted scatter: h[d] += w * xp[s] ----------
__global__ void k_edge_aggr(const int* __restrict__ EI, const float* __restrict__ Abuf,
                            const float* __restrict__ den, const float* __restrict__ XP,
                            float* __restrict__ Hb, int g0) {
  const int i = blockIdx.x, z = blockIdx.y, c = threadIdx.x;
  const int g = g0 + z;
  int s, d;
  if (i < E_) {
    s = EI[(long long)g * 2 * E_ + i];
    d = EI[(long long)g * 2 * E_ + E_ + i];
  } else {
    s = d = i - E_;
  }
  const long long nb = (long long)z * N_;
  const float w = Abuf[(long long)z * (E_ + N_) + i] / den[nb + d];
  atomicAdd(&Hb[(nb + d) * C_ + c], w * XP[(nb + s) * C_ + c]);
}

// ---------- bias + relu over [z,N,C] ----------
__global__ void k_bias_relu(float* __restrict__ Hb, const float* __restrict__ b,
                            long long total) {
  long long idx = (long long)blockIdx.x * blockDim.x + threadIdx.x;
  if (idx >= total) return;
  int c = (int)(idx % C_);
  Hb[idx] = fmaxf(Hb[idx] + b[c], 0.f);
}

// ---------- row L2 normalize (dim=1) ----------
__global__ void k_rownorm(float* __restrict__ Hb) {
  const int n = blockIdx.x, z = blockIdx.y, c = threadIdx.x;
  const long long base = ((long long)z * N_ + n) * C_;
  float v = Hb[base + c];
  __shared__ float s[C_];
  s[c] = v * v;
  __syncthreads();
  for (int off = 64; off > 0; off >>= 1) {
    if (c < off) s[c] += s[c + off];
    __syncthreads();
  }
  float norm = fmaxf(sqrtf(s[0]), 1e-12f);
  Hb[base + c] = v / norm;
}

// ---------- per-graph column standardization (ddof=1), in place ----------
__global__ void k_standardize(float* __restrict__ Z) {
  const int h = blockIdx.x * blockDim.x + threadIdx.x;   // 0..H-1
  const int z = blockIdx.y;
  float* col = Z + (long long)z * N_ * H_ + h;
  float sum = 0.f, sq = 0.f;
  for (int n = 0; n < N_; ++n) {
    float v = col[(long long)n * H_];
    sum += v; sq += v * v;
  }
  const float mean = sum / (float)N_;
  float var = (sq - (float)N_ * mean * mean) / (float)(N_ - 1);
  var = fmaxf(var, 0.f);
  const float stdv = sqrtf(var);
  for (int n = 0; n < N_; ++n) {
    long long ix = (long long)n * H_;
    col[ix] = (col[ix] - mean) / stdv;
  }
}

// ---------- launcher ----------
static inline unsigned ceil_div(long long a, long long b) { return (unsigned)((a + b - 1) / b); }

extern "C" void kernel_launch(void* const* d_in, const int* in_sizes, int n_in,
                              void* d_out, int out_size, void* d_ws, size_t ws_size,
                              hipStream_t stream) {
  const float* x[2]  = {(const float*)d_in[0], (const float*)d_in[3]};
  const int*   ei[2] = {(const int*)d_in[1],   (const int*)d_in[4]};
  const float* ea[2] = {(const float*)d_in[2], (const float*)d_in[5]};
  const float* W1 = (const float*)d_in[6];
  const float* a_src1 = (const float*)d_in[7];
  const float* a_dst1 = (const float*)d_in[8];
  const float* We1 = (const float*)d_in[9];
  const float* a_edge1 = (const float*)d_in[10];
  const float* b1 = (const float*)d_in[11];
  const float* W2 = (const float*)d_in[12];
  const float* a_src2 = (const float*)d_in[13];
  const float* a_dst2 = (const float*)d_in[14];
  const float* We2 = (const float*)d_in[15];
  const float* a_edge2 = (const float*)d_in[16];
  const float* b2 = (const float*)d_in[17];
  const float* Ws1 = (const float*)d_in[18];
  const float* bs1 = (const float*)d_in[19];
  const float* Ws2 = (const float*)d_in[20];
  const float* bs2 = (const float*)d_in[21];
  const float* Ws3 = (const float*)d_in[22];
  const float* bs3 = (const float*)d_in[23];
  float* out = (float*)d_out;

  // ---- workspace carve, chunked over graphs by available ws_size ----
  const long long fXP = (long long)N_ * C_;
  const long long fAb = (long long)(E_ + N_);
  const long long fS  = (long long)N_ * H_;
  const long long perGraphBytes = (2 * fXP + fAb + 6 * N_ + 2 * fS) * 4;
  long long avail = (long long)ws_size - 256;
  long long gchunk = avail > 0 ? avail / perGraphBytes : 1;
  if (gchunk < 1) gchunk = 1;
  if (gchunk > G_) gchunk = G_;

  char* p = (char*)d_ws;
  float* wedot = (float*)p;           p += 256;
  float* XP    = (float*)p;           p += gchunk * fXP * 4;
  float* Hb    = (float*)p;           p += gchunk * fXP * 4;
  float* Abuf  = (float*)p;           p += gchunk * fAb * 4;
  float* aS    = (float*)p;           p += gchunk * N_ * 4;
  float* aD    = (float*)p;           p += gchunk * N_ * 4;
  float* cnt   = (float*)p;           p += gchunk * N_ * 4;
  float* easum = (float*)p;           p += gchunk * N_ * 4;
  unsigned* am = (unsigned*)p;        p += gchunk * N_ * 4;
  float* den   = (float*)p;           p += gchunk * N_ * 4;
  float* S1    = (float*)p;           p += gchunk * fS * 4;
  float* S2    = (float*)p;           p += gchunk * fS * 4;

  k_wedot<<<1, C_, 0, stream>>>(We1, a_edge1, wedot, 0);
  k_wedot<<<1, C_, 0, stream>>>(We2, a_edge2, wedot, 1);

  for (int st = 0; st < 2; ++st) {
    for (int g0 = 0; g0 < G_; g0 += (int)gchunk) {
      const int ch = (int)((G_ - g0 < gchunk) ? (G_ - g0) : gchunk);
      const dim3 gEdge(ceil_div(E_, 256), ch);
      const dim3 gEN(ceil_div(E_ + N_, 256), ch);
      const dim3 gAggr(E_ + N_, ch);
      const dim3 gNode(N_, ch);

      // ===== GAT layer 1: xp = x @ W1 =====
      k_gemm_bf16<<<dim3(ceil_div(N_, BM), C_ / BN, ch), 256, 0, stream>>>(
          x[st] + (long long)g0 * N_ * F_, W1, nullptr, XP,
          N_, F_, C_, (long long)N_ * F_, fXP, 0);
      k_node_prep<<<gNode, C_, 0, stream>>>(XP, a_src1, a_dst1, aS, aD, cnt, easum, am, den, Hb);
      k_edge_stats<<<gEdge, 256, 0, stream>>>(ei[st], ea[st], cnt, easum, g0);
      k_edge_logit<<<gEN, 256, 0, stream>>>(ei[st], ea[st], aS, aD, cnt, easum, wedot, 0, Abuf, am, g0);
      k_edge_expsum<<<gEN, 256, 0, stream>>>(ei[st], Abuf, am, den, g0);
      k_edge_aggr<<<gAggr, C_, 0, stream>>>(ei[st], Abuf, den, XP, Hb, g0);
      k_bias_relu<<<ceil_div((long long)ch * fXP, 256), 256, 0, stream>>>(Hb, b1, (long long)ch * fXP);

      // ===== GAT layer 2: xp = h1 @ W2 =====
      k_gemm_bf16<<<dim3(ceil_div(N_, BM), C_ / BN, ch), 256, 0, stream>>>(
          Hb, W2, nullptr, XP, N_, C_, C_, fXP, fXP, 0);
      k_node_prep<<<gNode, C_, 0, stream>>>(XP, a_src2, a_dst2, aS, aD, cnt, easum, am, den, Hb);
      k_edge_stats<<<gEdge, 256, 0, stream>>>(ei[st], ea[st], cnt, easum, g0);
      k_edge_logit<<<gEN, 256, 0, stream>>>(ei[st], ea[st], aS, aD, cnt, easum, wedot, 1, Abuf, am, g0);
      k_edge_expsum<<<gEN, 256, 0, stream>>>(ei[st], Abuf, am, den, g0);
      k_edge_aggr<<<gAggr, C_, 0, stream>>>(ei[st], Abuf, den, XP, Hb, g0);
      k_bias_relu<<<ceil_div((long long)ch * fXP, 256), 256, 0, stream>>>(Hb, b2, (long long)ch * fXP);

      // ===== normalize rows =====
      k_rownorm<<<gNode, C_, 0, stream>>>(Hb);

      // ===== SOPOOL MLP (fused bias+relu WMMA GEMMs) =====
      k_gemm_bf16<<<dim3(ceil_div(N_, BM), H_ / BN, ch), 256, 0, stream>>>(
          Hb, Ws1, bs1, S1, N_, C_, H_, fXP, fS, 1);
      k_gemm_bf16<<<dim3(ceil_div(N_, BM), H_ / BN, ch), 256, 0, stream>>>(
          S1, Ws2, bs2, S2, N_, H_, H_, fS, fS, 1);
      float* outp = out + ((long long)st * G_ + g0) * N_ * H_;
      k_gemm_bf16<<<dim3(ceil_div(N_, BM), H_ / BN, ch), 256, 0, stream>>>(
          S2, Ws3, bs3, outp, N_, H_, H_, fS, fS, 1);

      // ===== per-graph column standardization (in place in d_out) =====
      k_standardize<<<dim3(H_ / 256, ch), 256, 0, stream>>>(outp);
    }
  }
}
